// MyNet_26225070309703
// MI455X (gfx1250) — matrix operand order repaired and checked by reading, not compiled, not run on hardware
//
#include <hip/hip_runtime.h>
#include <hip/hip_bf16.h>

typedef __attribute__((ext_vector_type(16))) _Float16 v16h;
typedef __attribute__((ext_vector_type(8)))  _Float16 v8h;
typedef __attribute__((ext_vector_type(8)))  float    v8f;

#define NROWS 8192      // b*n = 2*4096 points
#define NPB   4096      // points per batch
#define WID   16
#define HID   256
#define OUTF  4096      // WIDTH*WIDTH*RANK
#define IN_FEAT 13
#define BN_EPS 1e-5f

// ---------------------------------------------------------------- small stages

__global__ void fc0_kernel(const float* __restrict__ x, const float* __restrict__ w,
                           const float* __restrict__ b, float* __restrict__ v) {
    int idx = blockIdx.x * 256 + threadIdx.x;           // NROWS*WID
    int n = idx >> 4, o = idx & 15;
    float s = b[o];
    const float* xr = x + (size_t)n * IN_FEAT;
    const float* wr = w + (size_t)o * IN_FEAT;
    #pragma unroll
    for (int i = 0; i < IN_FEAT; ++i) s += xr[i] * wr[i];
    v[idx] = s;
}

// hidden = relu(v @ W1^T + b1), stored f16   (NROWS x HID)
__global__ void hidden_kernel(const float* __restrict__ v, const float* __restrict__ w1,
                              const float* __restrict__ b1, _Float16* __restrict__ hidden) {
    int idx = blockIdx.x * 256 + threadIdx.x;           // NROWS*HID
    int n = idx >> 8, h = idx & 255;
    float s = b1[h];
    const float* vr = v + (size_t)n * WID;
    const float* wr = w1 + (size_t)h * WID;
    #pragma unroll
    for (int i = 0; i < WID; ++i) s += vr[i] * wr[i];
    hidden[idx] = (_Float16)fmaxf(s, 0.0f);
}

__global__ void zero_kernel(float* __restrict__ p, int count) {
    int i = blockIdx.x * 256 + threadIdx.x;
    if (i < count) p[i] = 0.0f;
}

// ---------------------------------------------------------------- WMMA kernels

// psi GEMM (hidden[8192x256]f16 @ W2^T[256x4096]) fused with
// t[b, c] += (psi[n,c]) * v[n, i(c)]   (reduction over n)
// grid: 64 row-blocks (128 rows) x 256 col-blocks (16 cols)
__global__ __launch_bounds__(256)
void psi_accum_kernel(const _Float16* __restrict__ hidden,
                      const float* __restrict__ w2,    // [4096,256]
                      const float* __restrict__ b2,    // [4096]
                      const float* __restrict__ v,     // [8192,16]
                      float* __restrict__ t)           // [2,4096]
{
    __shared__ float t_lds[16];
    const int tid  = threadIdx.x;
    const int cb   = blockIdx.x & 255;
    const int rb   = blockIdx.x >> 8;
    const int wave = tid >> 5;
    const int lane = tid & 31;
    const int half = lane >> 4;
    const int lcol = lane & 15;
    const int cBase   = cb * 16;
    const int rowBase = rb * 128;
    const int batch   = rb >> 5;        // 32 row-blocks per batch

    if (tid < 16) t_lds[tid] = 0.0f;
    __syncthreads();

    const int c = cBase + lcol;
    // A: lane row M = lane%16 ; K runs at half*8 and 16+half*8 within each K-chunk
    const _Float16* arow = hidden + (size_t)(rowBase + wave * 16 + lcol) * HID;
    const float*    wrow = w2 + (size_t)c * HID;

    v8f acc = {};
    #pragma unroll
    for (int kc = 0; kc < 8; ++kc) {
        const int k0 = kc * 32 + half * 8;
        v8h alo = *(const v8h*)(arow + k0);
        v8h ahi = *(const v8h*)(arow + k0 + 16);
        v16h a;
        #pragma unroll
        for (int e = 0; e < 8; ++e) { a[e] = alo[e]; a[8 + e] = ahi[e]; }
        const float* wp = wrow + kc * 32 + half * 8;    // B: K at half*8 and 16+half*8, col N=lcol
        v16h b;
        #pragma unroll
        for (int e = 0; e < 16; ++e) b[e] = (_Float16)wp[e < 8 ? e : e + 8];
        acc = __builtin_amdgcn_wmma_f32_16x16x32_f16(false, a, false, b,
                                                     (short)0, acc, false, false);
    }
    // D: vgpr r -> row m = r + 8*half, col = lcol.  Fold in bias, multiply by v[n,i(c)].
    const float bias = b2[c];
    const int   ic   = (c >> 4) & 15;   // c = o*256 + i*16 + r
    float p = 0.0f;
    #pragma unroll
    for (int r = 0; r < 8; ++r) {
        int n = rowBase + wave * 16 + r + 8 * half;
        p += (acc[r] + bias) * v[(size_t)n * WID + ic];
    }
    atomicAdd(&t_lds[lcol], p);
    __syncthreads();
    if (tid < 16) atomicAdd(&t[(size_t)batch * OUTF + cBase + tid], t_lds[tid]);
}

// phi GEMM fused with contraction:  u[n,o] = sum_c t[b,c]*phi[n,c] + (v@Ww^T+bw)[n,o]
// t[c] is folded into the B operand so the WMMA accumulator reduces over c.
// Also accumulates BN batch statistics (sum, sumsq) per channel o.
// grid: 64 row-blocks x 16 output channels
__global__ __launch_bounds__(256)
void phi_contract_kernel(const _Float16* __restrict__ hidden,
                         const float* __restrict__ w2,    // phi l2 w [4096,256]
                         const float* __restrict__ b2,    // phi l2 b [4096]
                         const float* __restrict__ t,     // [2,4096]
                         const float* __restrict__ v,     // [8192,16]
                         const float* __restrict__ ww,    // [16,16]
                         const float* __restrict__ wb,    // [16]
                         float* __restrict__ u,           // [8192,16]
                         float* __restrict__ stats)       // [32] = sum[16], sq[16]
{
    __shared__ float s_lds[128];
    __shared__ float s_misc[3];   // [0]=bias term, [1]=sum, [2]=sumsq
    const int tid  = threadIdx.x;
    const int o    = blockIdx.x & 15;
    const int rb   = blockIdx.x >> 4;
    const int wave = tid >> 5;
    const int lane = tid & 31;
    const int half = lane >> 4;
    const int lcol = lane & 15;
    const int rowBase = rb * 128;
    const int batch   = rb >> 5;

    if (tid < 128) s_lds[tid] = 0.0f;
    if (tid < 3)   s_misc[tid] = 0.0f;
    __syncthreads();

    const float* tb = t + (size_t)batch * OUTF + o * 256;
    // phi bias contribution: sum_j t[c_j]*b2[c_j]  (constant per (batch,o))
    atomicAdd(&s_misc[0], tb[tid] * b2[o * 256 + tid]);

    const _Float16* arow = hidden + (size_t)(rowBase + wave * 16 + lcol) * HID;
    v16h a_all[8];
    #pragma unroll
    for (int kc = 0; kc < 8; ++kc) {
        const int k0 = kc * 32 + half * 8;
        v8h alo = *(const v8h*)(arow + k0);
        v8h ahi = *(const v8h*)(arow + k0 + 16);
        #pragma unroll
        for (int e = 0; e < 8; ++e) { a_all[kc][e] = alo[e]; a_all[kc][8 + e] = ahi[e]; }
    }

    v8f acc = {};
    for (int j = 0; j < 16; ++j) {                       // 16 col-chunks of the o-block
        const int   c  = o * 256 + j * 16 + lcol;
        const float tc = tb[j * 16 + lcol];
        const float* wrow = w2 + (size_t)c * HID;
        #pragma unroll
        for (int kc = 0; kc < 8; ++kc) {
            const float* wp = wrow + kc * 32 + half * 8;
            v16h b;
            #pragma unroll
            for (int e = 0; e < 16; ++e) b[e] = (_Float16)(wp[e < 8 ? e : e + 8] * tc);
            acc = __builtin_amdgcn_wmma_f32_16x16x32_f16(false, a_all[kc], false, b,
                                                         (short)0, acc, false, false);
        }
    }
    // acc[r]: partial sum over c for row m = wave*16 + r + 8*half; reduce 16 c-lanes via LDS
    #pragma unroll
    for (int r = 0; r < 8; ++r)
        atomicAdd(&s_lds[wave * 16 + r + 8 * half], acc[r]);
    __syncthreads();

    if (tid < 128) {
        const int n = rowBase + tid;
        float lin = wb[o];
        const float* vr = v + (size_t)n * WID;
        const float* wr = ww + (size_t)o * WID;
        #pragma unroll
        for (int i = 0; i < WID; ++i) lin += vr[i] * wr[i];
        const float uval = s_lds[tid] + s_misc[0] + lin;
        u[(size_t)n * WID + o] = uval;
        atomicAdd(&s_misc[1], uval);
        atomicAdd(&s_misc[2], uval * uval);
    }
    __syncthreads();
    if (tid == 0) {
        atomicAdd(&stats[o],      s_misc[1]);
        atomicAdd(&stats[16 + o], s_misc[2]);
    }
}

// BN (training-mode biased batch stats) + optional ReLU -> writes next v
__global__ void bn_kernel(const float* __restrict__ u, const float* __restrict__ stats,
                          const float* __restrict__ g, const float* __restrict__ bta,
                          float* __restrict__ v, int do_relu) {
    int idx = blockIdx.x * 256 + threadIdx.x;           // NROWS*WID
    int o = idx & 15;
    float mean = stats[o] * (1.0f / NROWS);
    float var  = stats[16 + o] * (1.0f / NROWS) - mean * mean;
    float xn = (u[idx] - mean) * rsqrtf(var + BN_EPS) * g[o] + bta[o];
    if (do_relu) xn = fmaxf(xn, 0.0f);
    v[idx] = xn;
}

// fc1: 16->128 + relu
__global__ void fc1_kernel(const float* __restrict__ v, const float* __restrict__ w,
                           const float* __restrict__ b, float* __restrict__ h1) {
    int idx = blockIdx.x * 256 + threadIdx.x;           // NROWS*128
    int n = idx >> 7, j = idx & 127;
    float s = b[j];
    const float* vr = v + (size_t)n * WID;
    const float* wr = w + (size_t)j * WID;
    #pragma unroll
    for (int i = 0; i < WID; ++i) s += vr[i] * wr[i];
    h1[idx] = fmaxf(s, 0.0f);
}

// fc2: 128->1
__global__ void fc2_kernel(const float* __restrict__ h1, const float* __restrict__ w,
                           const float* __restrict__ b, float* __restrict__ out) {
    int n = blockIdx.x * 256 + threadIdx.x;             // NROWS
    float s = b[0];
    const float* hr = h1 + (size_t)n * 128;
    #pragma unroll 8
    for (int j = 0; j < 128; ++j) s += hr[j] * w[j];
    out[n] = s;
}

// ---------------------------------------------------------------- host launcher

extern "C" void kernel_launch(void* const* d_in, const int* in_sizes, int n_in,
                              void* d_out, int out_size, void* d_ws, size_t ws_size,
                              hipStream_t stream) {
    // input map (depth-first dict insertion order)
    const float* x     = (const float*)d_in[0];
    const float* fc0w  = (const float*)d_in[1];
    const float* fc0b  = (const float*)d_in[2];
    // layer i base = 3 + i*12:
    //  +0 phi.l1.w  +1 phi.l1.b  +2 phi.l2.w  +3 phi.l2.b
    //  +4 psi.l1.w  +5 psi.l1.b  +6 psi.l2.w  +7 psi.l2.b
    //  +8 w.w  +9 w.b  +10 bn_g  +11 bn_b
    const float* fc1w  = (const float*)d_in[51];
    const float* fc1b  = (const float*)d_in[52];
    const float* fc2w  = (const float*)d_in[53];
    const float* fc2b  = (const float*)d_in[54];
    float* out = (float*)d_out;

    // workspace layout (byte offsets)
    char* ws = (char*)d_ws;
    float*    vbuf   = (float*)(ws + 0);                       // 8192*16  f32
    float*    ubuf   = (float*)(ws + 524288);                  // 8192*16  f32
    float*    tbuf   = (float*)(ws + 1048576);                 // 2*4096   f32
    float*    stats  = (float*)(ws + 1081344);                 // 32       f32 (contig. after t)
    float*    h1buf  = (float*)(ws + 1081600);                 // 8192*128 f32
    _Float16* hidden = (_Float16*)(ws + 5275904);              // 8192*256 f16

    // fc0
    fc0_kernel<<<(NROWS * WID) / 256, 256, 0, stream>>>(x, fc0w, fc0b, vbuf);

    for (int l = 0; l < 4; ++l) {
        const float* const* P = (const float* const*)(d_in + 3 + l * 12);
        const float *phi1w = P[0], *phi1b = P[1], *phi2w = P[2], *phi2b = P[3];
        const float *psi1w = P[4], *psi1b = P[5], *psi2w = P[6], *psi2b = P[7];
        const float *ww = P[8], *wb = P[9], *bng = P[10], *bnb = P[11];

        // psi hidden, zero t+stats, psi GEMM + t accumulation
        hidden_kernel<<<(NROWS * HID) / 256, 256, 0, stream>>>(vbuf, psi1w, psi1b, hidden);
        zero_kernel<<<(2 * OUTF + 32 + 255) / 256, 256, 0, stream>>>(tbuf, 2 * OUTF + 32);
        psi_accum_kernel<<<64 * 256, 256, 0, stream>>>(hidden, psi2w, psi2b, vbuf, tbuf);

        // phi hidden, phi GEMM + contraction + linear + BN stats
        hidden_kernel<<<(NROWS * HID) / 256, 256, 0, stream>>>(vbuf, phi1w, phi1b, hidden);
        phi_contract_kernel<<<64 * 16, 256, 0, stream>>>(hidden, phi2w, phi2b, tbuf,
                                                         vbuf, ww, wb, ubuf, stats);
        // BN (+ReLU except last layer), writes next v
        bn_kernel<<<(NROWS * WID) / 256, 256, 0, stream>>>(ubuf, stats, bng, bnb,
                                                           vbuf, (l < 3) ? 1 : 0);
    }

    fc1_kernel<<<(NROWS * 128) / 256, 256, 0, stream>>>(vbuf, fc1w, fc1b, h1buf);
    fc2_kernel<<<NROWS / 256, 256, 0, stream>>>(h1buf, fc2w, fc2b, out);
    (void)in_sizes; (void)n_in; (void)out_size; (void)ws_size;
}
